// ReceiverModel_27298812134053
// MI455X (gfx1250) — matrix operand order repaired and checked by reading, not compiled, not run on hardware
//
#include <hip/hip_runtime.h>

// ---------------- problem constants (from reference) ----------------
constexpr int Nn   = 16384;   // nodes
constexpr int Ne   = 131072;  // edges
constexpr int FINc = 16;
constexpr int HIDc = 128;
constexpr int Hc   = 4;
constexpr int DHc  = 32;
constexpr int NV   = 4;       // D2 views
constexpr int M4   = NV * Nn; // 65536 batched rows (view-major)
constexpr float NEG_SLOPE = 0.2f;

typedef __bf16 bf16_t;
typedef __attribute__((ext_vector_type(16))) __bf16 v16bf;
typedef __attribute__((ext_vector_type(8)))  float  v8f;

// ---------------- helpers ----------------
__device__ __forceinline__ unsigned fkey(float f) {
  unsigned b = __float_as_uint(f);
  return (b & 0x80000000u) ? ~b : (b | 0x80000000u);
}
__device__ __forceinline__ float funkey(unsigned k) {
  unsigned b = (k & 0x80000000u) ? (k ^ 0x80000000u) : ~k;
  return __uint_as_float(b);
}

// ---------------- kernels ----------------

// x [N,16] -> xv bf16 [4N, 32] (K zero-padded to 32), sign flips on cols 0,1
__global__ void build_views(const float* __restrict__ x, bf16_t* __restrict__ xv) {
  int t = blockIdx.x * blockDim.x + threadIdx.x;   // one thread per (v,n)
  if (t >= M4) return;
  int n = t % Nn, v = t / Nn;
  float sx = (v == 1 || v == 3) ? -1.f : 1.f;
  float sy = (v == 2 || v == 3) ? -1.f : 1.f;
  const float* xi = x + (size_t)n * FINc;
  bf16_t* o = xv + (size_t)t * 32;
#pragma unroll
  for (int k = 0; k < FINc; ++k) {
    float f = xi[k];
    if (k == 0) f *= sx;
    if (k == 1) f *= sy;
    o[k] = (bf16_t)f;
  }
#pragma unroll
  for (int k = FINc; k < 32; ++k) o[k] = (bf16_t)0.f;
}

// f32 W[K,Nc] -> bf16 transposed Wt[Nc,Kpad] (cols k >= K are zero)
__global__ void cvt_weight_t(const float* __restrict__ in, bf16_t* __restrict__ out,
                             int K, int Kpad, int Nc) {
  int t = blockIdx.x * blockDim.x + threadIdx.x;   // t = n*Kpad + k
  if (t >= Nc * Kpad) return;
  int n = t / Kpad, k = t % Kpad;
  out[t] = (k < K) ? (bf16_t)in[(size_t)k * Nc + n] : (bf16_t)0.f;
}

__global__ void cvt_bf16(const float* __restrict__ in, bf16_t* __restrict__ out, int n) {
  int t = blockIdx.x * blockDim.x + threadIdx.x;
  if (t < n) out[t] = (bf16_t)in[t];
}

__global__ void fill_u32(unsigned* __restrict__ p, unsigned v, int n) {
  int t = blockIdx.x * blockDim.x + threadIdx.x;
  if (t < n) p[t] = v;
}

// One wave computes a 64x16 supertile (4 stacked 16x16 WMMA tiles) of
// C = A[M,K](bf16) @ B[K,Nout](bf16), with B given transposed Bt[Nout,K].
// KSTEPS = K/32 is a compile-time constant -> fully unrolled WMMA stream.
// B tile loaded once per k-step and reused by 4 independent WMMAs.
template <int KSTEPS, bool BIAS, bool RELU>
__global__ void gemm64_bf16(const bf16_t* __restrict__ A, const bf16_t* __restrict__ Bt,
                            float* __restrict__ C, int M, int Nout,
                            const float* __restrict__ bias) {
  constexpr int K = KSTEPS * 32;
  const int wave = (blockIdx.x * blockDim.x + threadIdx.x) >> 5;
  const int lane = threadIdx.x & 31;
  const int tilesN = Nout >> 4;
  const int tileM = wave / tilesN;      // 64-row supertile index
  const int tileN = wave % tilesN;
  if (tileM * 64 >= M) return;          // wave-uniform exit, EXEC stays all-ones

  const int half  = lane >> 4;          // 0 or 1
  const int idx16 = lane & 15;
  const int bcol  = tileN * 16 + idx16; // B/C column for this lane

  v8f acc[4] = {v8f{}, v8f{}, v8f{}, v8f{}};
  const bf16_t* Bp0 = Bt + (size_t)bcol * K;

#pragma unroll
  for (int ks = 0; ks < KSTEPS; ++ks) {
    const int k0 = ks * 32;
    // B: lanes 0-15 hold K=k0..k0+15, lanes 16-31 hold K=k0+16..k0+31 (contiguous)
    v16bf b;
    const bf16_t* Bp = Bp0 + k0 + half * 16;
#pragma unroll
    for (int e = 0; e < 16; ++e) b[e] = Bp[e];

#pragma unroll
    for (int mt = 0; mt < 4; ++mt) {
      const int arow = tileM * 64 + mt * 16 + idx16;
      const bf16_t* Ap = A + (size_t)arow * K + k0;
      v16bf a;
#pragma unroll
      for (int j = 0; j < 8; ++j) {
        // lanes 0-15: K = 0..15; lanes 16-31: K = 16..31 (contiguous)
        int kk = half * 16 + (j << 1);
        a[2 * j]     = Ap[kk];
        a[2 * j + 1] = Ap[kk + 1];
      }
      acc[mt] = __builtin_amdgcn_wmma_f32_16x16x32_bf16(false, a, false, b,
                                                        (short)0, acc[mt], false, false);
    }
  }

#pragma unroll
  for (int mt = 0; mt < 4; ++mt) {
#pragma unroll
    for (int j = 0; j < 8; ++j) {
      int row = tileM * 64 + mt * 16 + half * 8 + j;  // D: VGPR j -> M = half*8 + j
      float v = acc[mt][j];
      if constexpr (BIAS) v += bias[bcol];
      if constexpr (RELU) v = fmaxf(v, 0.f);
      C[(size_t)row * Nout + bcol] = v;
    }
  }
}

// per (view, edge, head): e = a . leaky_relu(hl[src] + hr[dst]); atomic segment-max
__global__ void edge_scores(const float* __restrict__ hl, const float* __restrict__ hr,
                            const float* __restrict__ avec,
                            const int* __restrict__ src, const int* __restrict__ dst,
                            float* __restrict__ e, unsigned* __restrict__ mkeys) {
  int t = blockIdx.x * blockDim.x + threadIdx.x;   // t = ((v*Ne + eidx)*Hc + h)
  if (t >= NV * Ne * Hc) return;
  int h    = t % Hc;
  int eidx = (t / Hc) % Ne;
  int v    = t / (Hc * Ne);
  int s = src[eidx], d = dst[eidx];
  const float* pl = hl + ((size_t)(v * Nn + s)) * HIDc + h * DHc;
  const float* pr = hr + ((size_t)(v * Nn + d)) * HIDc + h * DHc;
  const float* pa = avec + h * DHc;
  float accv = 0.f;
#pragma unroll
  for (int k = 0; k < DHc; ++k) {
    float xv = pl[k] + pr[k];
    xv = xv > 0.f ? xv : NEG_SLOPE * xv;
    accv += xv * pa[k];
  }
  e[t] = accv;
  atomicMax(&mkeys[((size_t)(v * Nn + d)) * Hc + h], fkey(accv));
}

// per (view, edge, head): ex = exp(e - max); denom += ex; acc[dst] += ex * hl[src]
__global__ void edge_accum(const float* __restrict__ hl, const float* __restrict__ e,
                           const unsigned* __restrict__ mkeys,
                           const int* __restrict__ src, const int* __restrict__ dst,
                           float* __restrict__ denom, float* __restrict__ acc) {
  int t = blockIdx.x * blockDim.x + threadIdx.x;
  if (t >= NV * Ne * Hc) return;
  int h    = t % Hc;
  int eidx = (t / Hc) % Ne;
  int v    = t / (Hc * Ne);
  int s = src[eidx], d = dst[eidx];
  float m  = funkey(mkeys[((size_t)(v * Nn + d)) * Hc + h]);
  float ex = __expf(e[t] - m);
  atomicAdd(&denom[((size_t)(v * Nn + d)) * Hc + h], ex);
  const float* pl = hl + ((size_t)(v * Nn + s)) * HIDc + h * DHc;
  float*       po = acc + ((size_t)(v * Nn + d)) * HIDc + h * DHc;
#pragma unroll
  for (int k = 0; k < DHc; ++k) atomicAdd(&po[k], ex * pl[k]);
}

// h = (act? elu : id)(acc/(denom+1e-16) + res), elementwise over [4N, HID]
__global__ void node_finish(const float* __restrict__ acc, const float* __restrict__ denom,
                            const float* __restrict__ res, float* __restrict__ hout, int act) {
  int t = blockIdx.x * blockDim.x + threadIdx.x;
  if (t >= M4 * HIDc) return;
  int f = t % HIDc;
  int row = t / HIDc;
  int h = f / DHc;
  float v = acc[t] / (denom[(size_t)row * Hc + h] + 1e-16f) + res[t];
  if (act) v = v > 0.f ? v : (__expf(v) - 1.f);
  hout[t] = v;
}

// emb_bf16[n,f] = mean over views of h[v*N+n, f]
__global__ void view_mean(const float* __restrict__ h, bf16_t* __restrict__ embbf) {
  int t = blockIdx.x * blockDim.x + threadIdx.x;  // over N*HID
  if (t >= Nn * HIDc) return;
  int f = t % HIDc, n = t / HIDc;
  float s = 0.f;
#pragma unroll
  for (int v = 0; v < NV; ++v) s += h[((size_t)(v * Nn + n)) * HIDc + f];
  embbf[t] = (bf16_t)(s * 0.25f);
}

// logits[n] = h1[n,:] . Wh2 + bh2
__global__ void head_out(const float* __restrict__ h1, const float* __restrict__ Wh2,
                         const float* __restrict__ bh2, float* __restrict__ out) {
  int n = blockIdx.x * blockDim.x + threadIdx.x;
  if (n >= Nn) return;
  float s = bh2[0];
#pragma unroll 8
  for (int k = 0; k < HIDc; ++k) s += h1[(size_t)n * HIDc + k] * Wh2[k];
  out[n] = s;
}

// ---------------- host orchestration ----------------
extern "C" void kernel_launch(void* const* d_in, const int* in_sizes, int n_in,
                              void* d_out, int out_size, void* d_ws, size_t ws_size,
                              hipStream_t stream) {
  (void)in_sizes; (void)n_in; (void)out_size; (void)ws_size;
  const float* x     = (const float*)d_in[0];
  const int*   ei    = (const int*)d_in[1];
  const int*   srcp  = ei;
  const int*   dstp  = ei + Ne;
  const float* W0l   = (const float*)d_in[3];
  const float* W0r   = (const float*)d_in[4];
  const float* a0    = (const float*)d_in[5];
  const float* Wres0 = (const float*)d_in[6];
  const float* Wl    = (const float*)d_in[7];
  const float* Wr    = (const float*)d_in[8];
  const float* att   = (const float*)d_in[9];
  const float* Wh1   = (const float*)d_in[10];
  const float* bh1   = (const float*)d_in[11];
  const float* Wh2   = (const float*)d_in[12];
  const float* bh2   = (const float*)d_in[13];

  char* ws = (char*)d_ws;
  size_t off = 0;
  auto take = [&](size_t bytes) -> char* {
    char* p = ws + off;
    off += (bytes + 255) & ~(size_t)255;
    return p;
  };
  bf16_t*   hbf   = (bf16_t*)take((size_t)M4 * HIDc * 2); // also holds XV / emb_bf16
  float*    h     = (float*)take((size_t)M4 * HIDc * 4);
  float*    hl    = (float*)take((size_t)M4 * HIDc * 4);
  float*    hr    = (float*)take((size_t)M4 * HIDc * 4); // also head hidden h1
  float*    acc   = (float*)take((size_t)M4 * HIDc * 4);
  float*    res   = (float*)take((size_t)M4 * HIDc * 4);
  float*    ebuf  = (float*)take((size_t)NV * Ne * Hc * 4);
  unsigned* mk    = (unsigned*)take((size_t)M4 * Hc * 4);
  float*    denom = (float*)take((size_t)M4 * Hc * 4);
  bf16_t* w0l_b  = (bf16_t*)take(128 * 32 * 2);   // transposed [N=128, Kpad=32]
  bf16_t* w0r_b  = (bf16_t*)take(128 * 32 * 2);
  bf16_t* wres_b = (bf16_t*)take(128 * 32 * 2);
  bf16_t* wl_b0  = (bf16_t*)take(128 * 128 * 2);  // transposed [128,128]
  bf16_t* wl_b1  = (bf16_t*)take(128 * 128 * 2);
  bf16_t* wr_b0  = (bf16_t*)take(128 * 128 * 2);
  bf16_t* wr_b1  = (bf16_t*)take(128 * 128 * 2);
  bf16_t* wh1_b  = (bf16_t*)take(128 * 128 * 2);

  const int TB = 256;
  // --- setup: views + weight conversion (transposed, K zero-padded) ---
  build_views<<<(M4 + TB - 1) / TB, TB, 0, stream>>>(x, hbf);
  cvt_weight_t<<<(128 * 32 + TB - 1) / TB, TB, 0, stream>>>(W0l, w0l_b, FINc, 32, HIDc);
  cvt_weight_t<<<(128 * 32 + TB - 1) / TB, TB, 0, stream>>>(W0r, w0r_b, FINc, 32, HIDc);
  cvt_weight_t<<<(128 * 32 + TB - 1) / TB, TB, 0, stream>>>(Wres0, wres_b, FINc, 32, HIDc);
  cvt_weight_t<<<(128 * 128 + TB - 1) / TB, TB, 0, stream>>>(Wl,             wl_b0, 128, 128, 128);
  cvt_weight_t<<<(128 * 128 + TB - 1) / TB, TB, 0, stream>>>(Wl + 128 * 128, wl_b1, 128, 128, 128);
  cvt_weight_t<<<(128 * 128 + TB - 1) / TB, TB, 0, stream>>>(Wr,             wr_b0, 128, 128, 128);
  cvt_weight_t<<<(128 * 128 + TB - 1) / TB, TB, 0, stream>>>(Wr + 128 * 128, wr_b1, 128, 128, 128);
  cvt_weight_t<<<(128 * 128 + TB - 1) / TB, TB, 0, stream>>>(Wh1, wh1_b, 128, 128, 128);

  // 64-row supertiles: blocks = (M/64)*(Nout/16) waves / 8 waves-per-block
  const int gemmBlocksBig  = (M4 / 64) * (HIDc / 16) / 8;  // 1024
  const int gemmBlocksHead = (Nn / 64) * (HIDc / 16) / 8;  // 256
  const int edgeThreads = NV * Ne * Hc;
  const int nodeThreads = M4 * HIDc;

  // --- layer 0 (K padded 16->32, KSTEPS=1) ---
  gemm64_bf16<1, false, false><<<gemmBlocksBig, TB, 0, stream>>>(hbf, w0l_b, hl, M4, HIDc, nullptr);
  gemm64_bf16<1, false, false><<<gemmBlocksBig, TB, 0, stream>>>(hbf, w0r_b, hr, M4, HIDc, nullptr);
  gemm64_bf16<1, false, false><<<gemmBlocksBig, TB, 0, stream>>>(hbf, wres_b, res, M4, HIDc, nullptr);
  fill_u32<<<(M4 * Hc + TB - 1) / TB, TB, 0, stream>>>(mk, 0u, M4 * Hc);
  fill_u32<<<(M4 * Hc + TB - 1) / TB, TB, 0, stream>>>((unsigned*)denom, 0u, M4 * Hc);
  fill_u32<<<(nodeThreads + TB - 1) / TB, TB, 0, stream>>>((unsigned*)acc, 0u, nodeThreads);
  edge_scores<<<(edgeThreads + TB - 1) / TB, TB, 0, stream>>>(hl, hr, a0, srcp, dstp, ebuf, mk);
  edge_accum<<<(edgeThreads + TB - 1) / TB, TB, 0, stream>>>(hl, ebuf, mk, srcp, dstp, denom, acc);
  node_finish<<<(nodeThreads + TB - 1) / TB, TB, 0, stream>>>(acc, denom, res, h, 1);

  // --- layers 1..2 (K=128, KSTEPS=4) ---
  for (int i = 0; i < 2; ++i) {
    bf16_t* wlb = (i == 0) ? wl_b0 : wl_b1;
    bf16_t* wrb = (i == 0) ? wr_b0 : wr_b1;
    cvt_bf16<<<(nodeThreads + TB - 1) / TB, TB, 0, stream>>>(h, hbf, nodeThreads);
    gemm64_bf16<4, false, false><<<gemmBlocksBig, TB, 0, stream>>>(hbf, wlb, hl, M4, HIDc, nullptr);
    gemm64_bf16<4, false, false><<<gemmBlocksBig, TB, 0, stream>>>(hbf, wrb, hr, M4, HIDc, nullptr);
    fill_u32<<<(M4 * Hc + TB - 1) / TB, TB, 0, stream>>>(mk, 0u, M4 * Hc);
    fill_u32<<<(M4 * Hc + TB - 1) / TB, TB, 0, stream>>>((unsigned*)denom, 0u, M4 * Hc);
    fill_u32<<<(nodeThreads + TB - 1) / TB, TB, 0, stream>>>((unsigned*)acc, 0u, nodeThreads);
    edge_scores<<<(edgeThreads + TB - 1) / TB, TB, 0, stream>>>(hl, hr, att + i * Hc * DHc,
                                                                srcp, dstp, ebuf, mk);
    edge_accum<<<(edgeThreads + TB - 1) / TB, TB, 0, stream>>>(hl, ebuf, mk, srcp, dstp, denom, acc);
    node_finish<<<(nodeThreads + TB - 1) / TB, TB, 0, stream>>>(acc, denom, h, h, (i == 0) ? 1 : 0);
  }

  // --- head: mean over views -> Wh1 (+bias, relu) -> Wh2 + bh2 ---
  view_mean<<<(Nn * HIDc + TB - 1) / TB, TB, 0, stream>>>(h, hbf);
  gemm64_bf16<4, true, true><<<gemmBlocksHead, TB, 0, stream>>>(hbf, wh1_b, hr, Nn, HIDc, bh1);
  head_out<<<(Nn + TB - 1) / TB, TB, 0, stream>>>(hr, Wh2, bh2, (float*)d_out);
}
